// GAT_17128329577087
// MI455X (gfx1250) — compile-verified, matrix operand order checked by reading, not run on hardware
//
#include <hip/hip_runtime.h>
#include <hip/hip_bf16.h>

#define N_NODES 6000
#define JPAD    6016
#define IN_DIMK 128
#define NH      8
#define NF      64
#define C_DIM   512
#define NEGV    (-9e15f)
#define NCH     188          // ceil(6000/32) chunks of 32 target nodes

typedef float  v2f   __attribute__((ext_vector_type(2)));
typedef float  v8f   __attribute__((ext_vector_type(8)));
typedef __bf16 v16bf __attribute__((ext_vector_type(16)));
typedef unsigned short us8 __attribute__((ext_vector_type(8)));

static __device__ __forceinline__ unsigned short f2bfbits(float f) {
  union { float f; unsigned u; } c; c.f = f;
  return (unsigned short)((c.u + 0x7fffu + ((c.u >> 16) & 1u)) >> 16);  // RNE
}
static __device__ __forceinline__ __bf16 f2bf(float f) {
  union { unsigned short s; __bf16 b; } o; o.s = f2bfbits(f); return o.b;
}
// partner-lane exchange across the two 16-lane halves (SWAPX16: and=0x1f, xor=0x10)
static __device__ __forceinline__ float swz_xor16(float v) {
  return __int_as_float(__builtin_amdgcn_ds_swizzle(__float_as_int(v), 0x401f));
}

// ---------------------------------------------------------------------------
// Kernel 1: fp32 GEMM  Out[6000,512] = X[6000,128] @ Wm[128,512]
// exact fp32 via V_WMMA_F32_16X16X4_F32, one 16x16 tile per wave, K loop of 32.
// ---------------------------------------------------------------------------
__global__ __launch_bounds__(128) void gat_proj_gemm(
    const float* __restrict__ X, const float* __restrict__ Wm,
    float* __restrict__ Out) {
  const int lane = threadIdx.x;          // 0..31
  const int wave = threadIdx.y;          // 0..3
  const int row  = lane & 15;
  const int half = lane >> 4;
  const int n0   = blockIdx.x * 64 + wave * 16;
  const int i0   = blockIdx.y * 16;
  const int col  = row;

  // A 16x4 f32: lanes 0-15 hold (M=row, K=k..k+1); lanes 16-31 (M=row, K=k+2..k+3)
  const float* xrow = X + (size_t)(i0 + row) * IN_DIMK + half * 2;
  v8f c = {};
  #pragma unroll 4
  for (int k = 0; k < IN_DIMK; k += 4) {
    v2f a, b;
    a.x = xrow[k];
    a.y = xrow[k + 1];
    const int kb = k + half * 2;
    b.x = Wm[(size_t)kb * C_DIM + n0 + col];
    b.y = Wm[(size_t)(kb + 1) * C_DIM + n0 + col];
    c = __builtin_amdgcn_wmma_f32_16x16x4_f32(false, a, false, b, (short)0, c,
                                              false, false);
  }
  #pragma unroll
  for (int r = 0; r < 8; ++r)
    Out[(size_t)(i0 + half * 8 + r) * C_DIM + n0 + col] = c[r];
}

// ---------------------------------------------------------------------------
// Kernel 2: per-head attention logits  s_src[h][n], s_tgt[h][n] (s_tgt padded)
// ---------------------------------------------------------------------------
__global__ void gat_scores(const float* __restrict__ proj,
                           const float* __restrict__ a_src,
                           const float* __restrict__ a_tgt,
                           float* __restrict__ s_src,
                           float* __restrict__ s_tgt) {
  int gid = blockIdx.x * blockDim.x + threadIdx.x;
  if (gid >= NH * JPAD) return;
  int h = gid / JPAD, j = gid - h * JPAD;
  if (j >= N_NODES) { s_tgt[h * JPAD + j] = 0.f; return; }
  const float* pr = proj + (size_t)j * C_DIM + h * NF;
  const float* as = a_src + h * NF;
  const float* at = a_tgt + h * NF;
  float ss = 0.f, st = 0.f;
  #pragma unroll 8
  for (int f = 0; f < NF; ++f) { float p = pr[f]; ss += p * as[f]; st += p * at[f]; }
  s_src[h * N_NODES + j] = ss;
  s_tgt[h * JPAD + j]    = st;
}

// ---------------------------------------------------------------------------
// Kernel 3: LDS-tiled transpose  projT[h*64+f][j] = bf16(proj[j][h*64+f])
// coalesced reads and writes; zero-padded to JPAD columns.
// ---------------------------------------------------------------------------
__global__ __launch_bounds__(256) void gat_transpose_bf16(
    const float* __restrict__ proj, __bf16* __restrict__ projT) {
  __shared__ float t[32][33];
  const int j0 = blockIdx.x * 32;
  const int c0 = blockIdx.y * 32;
  const int tx = threadIdx.x;            // 0..31
  const int ty = threadIdx.y;            // 0..7
  #pragma unroll
  for (int k = 0; k < 4; ++k) {
    const int j = j0 + ty + k * 8;
    t[ty + k * 8][tx] = (j < N_NODES) ? proj[(size_t)j * C_DIM + c0 + tx] : 0.f;
  }
  __syncthreads();
  #pragma unroll
  for (int k = 0; k < 4; ++k) {
    const int ch = c0 + ty + k * 8;
    projT[(size_t)ch * JPAD + j0 + tx] = f2bf(t[tx][ty + k * 8]);
  }
}

// ---------------------------------------------------------------------------
// Kernel 4: fused masked-softmax aggregation (flash style).
// Block = 8 waves = 8 heads, same 16-row i-tile.
// adj tiles (16x32 f32) streamed Global->LDS once per block with
// GLOBAL_LOAD_ASYNC_TO_LDS_B64, 4-deep pipeline on ASYNCcnt.
// Each wave: 4 WMMA accumulators (16x16 f32) covering F=64, bf16 K=32 chunks.
// ---------------------------------------------------------------------------
#define ADJ_STRIDE 36      // 36 floats/row: rows land on distinct bank groups

__global__ __launch_bounds__(256) void gat_flash_aggregate(
    const float* __restrict__ adj, const float* __restrict__ s_src,
    const float* __restrict__ s_tgt, const __bf16* __restrict__ projT,
    const float* __restrict__ skipb, const float* __restrict__ bias,
    float* __restrict__ out) {
  __shared__ float adjT[4][16][ADJ_STRIDE];   // 4-deep pipeline, 9216 B
  __shared__ float lbuf[NH][16];

  const int lane = threadIdx.x;        // 0..31
  const int h    = threadIdx.y;        // head = wave
  const int row  = lane & 15;          // M index within tile (both halves)
  const int half = lane >> 4;          // 0: K {0..7,16..23}  1: K {8..15,24..31}
  const int i0   = blockIdx.x * 16;
  const int i    = i0 + row;

  // cooperative copy mapping: 256 threads x one f32-pair of the 16x32 tile
  const int tid  = h * 32 + lane;      // 0..255
  const int crow = tid >> 4;           // 0..15
  const int ccol = (tid & 15) * 2;     // 0,2,..,30 (pairs never straddle 6000)
  const unsigned rowBase = (unsigned)(i0 + crow) * (unsigned)N_NODES;
  const unsigned long long adjAddr = (unsigned long long)adj;

  // issue one async Global->LDS b64 copy of this thread's pair of tile `t`
  auto issue_tile = [&](int t) {
    const int jc = t * 32;
    const unsigned loff =
        (unsigned)(unsigned long long)(const void*)&adjT[t & 3][crow][ccol];
    if (jc + ccol < N_NODES) {
      const unsigned voff = (rowBase + (unsigned)(jc + ccol)) * 4u;
      asm volatile("global_load_async_to_lds_b64 %0, %1, %2"
                   :: "v"(loff), "v"(voff), "s"(adjAddr)
                   : "memory");
    } else {                           // zero-fill OOB tail (last chunk only)
      adjT[t & 3][crow][ccol]     = 0.f;
      adjT[t & 3][crow][ccol + 1] = 0.f;
    }
  };

  const float  ssrc   = s_src[h * N_NODES + i];
  const float* stg    = s_tgt + h * JPAD;
  const __bf16* bbase = projT + (size_t)(h * NF) * JPAD;

  v8f c[4] = {{}, {}, {}, {}};
  float m = -__builtin_inff(), l = 0.f;

  issue_tile(0); issue_tile(1); issue_tile(2);   // prime the pipeline

  for (int n = 0; n < NCH; ++n) {
    const int cur = n & 3;
    if (n + 3 < NCH) issue_tile(n + 3);

    // wait until tile n is resident (allow newer tiles to stay in flight)
    const int rem = (NCH - 1) - n;     // tiles newer than n still outstanding
    if (rem >= 3)      asm volatile("s_wait_asynccnt 3" ::: "memory");
    else if (rem == 2) asm volatile("s_wait_asynccnt 2" ::: "memory");
    else if (rem == 1) asm volatile("s_wait_asynccnt 1" ::: "memory");
    else               asm volatile("s_wait_asynccnt 0" ::: "memory");
    __syncthreads();                   // all waves' copy portions complete

    const int jc = n * 32;
    const int r0 = jc + half * 8;
    const int r1 = jc + 16 + half * 8;

    // adjacency mask from LDS tile (bank-friendly padded rows)
    const float* t0 = &adjT[cur][row][half * 8];
    const float* t1 = &adjT[cur][row][16 + half * 8];
    float4 a01 = *(const float4*)(t0);
    float4 a02 = *(const float4*)(t0 + 4);
    float4 a11 = *(const float4*)(t1);
    float4 a12 = *(const float4*)(t1 + 4);

    // target logits (padded buffer, always in bounds)
    float4 t01 = *(const float4*)(stg + r0);
    float4 t02 = *(const float4*)(stg + r0 + 4);
    float4 t11 = *(const float4*)(stg + r1);
    float4 t12 = *(const float4*)(stg + r1 + 4);

    const float adjv[16] = {a01.x,a01.y,a01.z,a01.w, a02.x,a02.y,a02.z,a02.w,
                            a11.x,a11.y,a11.z,a11.w, a12.x,a12.y,a12.z,a12.w};
    const float tv[16]   = {t01.x,t01.y,t01.z,t01.w, t02.x,t02.y,t02.z,t02.w,
                            t11.x,t11.y,t11.z,t11.w, t12.x,t12.y,t12.z,t12.w};

    float sc[16];
    #pragma unroll
    for (int k = 0; k < 16; ++k) {
      float s = ssrc + tv[k];
      s = (s > 0.f) ? s : 0.2f * s;                 // leaky_relu(0.2)
      sc[k] = s + NEGV * (1.f - adjv[k]);           // mask non-edges
    }

    float cmax = sc[0];
    #pragma unroll
    for (int k = 1; k < 16; ++k) cmax = fmaxf(cmax, sc[k]);
    cmax = fmaxf(cmax, swz_xor16(cmax));            // combine row halves
    const float mnew  = fmaxf(m, cmax);
    const float scale = __expf(m - mnew);

    float p[16], ls = 0.f;
    #pragma unroll
    for (int k = 0; k < 16; ++k) { p[k] = __expf(sc[k] - mnew); ls += p[k]; }
    ls += swz_xor16(ls);
    l = l * scale + ls;
    m = mnew;

    // broadcast per-row rescale factors through LDS (same-wave DS is in-order)
    if (half == 0) lbuf[h][row] = scale;
    float sr[8];
    #pragma unroll
    for (int r = 0; r < 8; ++r) sr[r] = lbuf[h][half * 8 + r];
    #pragma unroll
    for (int r = 0; r < 8; ++r) {
      c[0][r] *= sr[r]; c[1][r] *= sr[r]; c[2][r] *= sr[r]; c[3][r] *= sr[r];
    }

    // A fragment: 16x32 bf16 probabilities (elements = K runs, ascending)
    union { unsigned short s[16]; v16bf v; } Af;
    #pragma unroll
    for (int k = 0; k < 16; ++k) Af.s[k] = f2bfbits(p[k]);

    // B fragments: projT column slices, b128 runs; accumulate 4 F-slices
    #pragma unroll
    for (int f4 = 0; f4 < 4; ++f4) {
      const __bf16* bcol = bbase + (size_t)(f4 * 16 + row) * JPAD;
      union { us8 q[2]; v16bf v; } Bf;
      Bf.q[0] = *(const us8*)(bcol + r0);
      Bf.q[1] = *(const us8*)(bcol + r1);
      c[f4] = __builtin_amdgcn_wmma_f32_16x16x32_bf16(
          false, Af.v, false, Bf.v, (short)0, c[f4], false, false);
    }

    __syncthreads();   // everyone done reading adjT[cur] before it is reused
  }

  // normalize, add projected skip + bias, output leaky_relu
  if (half == 0) lbuf[h][row] = l;
  float li[8];
  #pragma unroll
  for (int r = 0; r < 8; ++r) li[r] = 1.f / lbuf[h][half * 8 + r];

  #pragma unroll
  for (int f4 = 0; f4 < 4; ++f4) {
    #pragma unroll
    for (int r = 0; r < 8; ++r) {
      const int rr = i0 + half * 8 + r;
      const int ch = h * NF + f4 * 16 + row;
      float v = c[f4][r] * li[r] + skipb[(size_t)rr * C_DIM + ch] + bias[ch];
      out[(size_t)rr * C_DIM + ch] = (v > 0.f) ? v : 0.2f * v;
    }
  }
}

// ---------------------------------------------------------------------------
extern "C" void kernel_launch(void* const* d_in, const int* in_sizes, int n_in,
                              void* d_out, int out_size, void* d_ws,
                              size_t ws_size, hipStream_t stream) {
  const float* x     = (const float*)d_in[0];
  const float* adj   = (const float*)d_in[1];
  const float* Wm    = (const float*)d_in[2];
  const float* a_src = (const float*)d_in[3];
  const float* a_tgt = (const float*)d_in[4];
  const float* skW   = (const float*)d_in[5];
  const float* bias  = (const float*)d_in[6];
  float* out = (float*)d_out;

  char* ws = (char*)d_ws;
  float* proj  = (float*)ws;  ws += (size_t)N_NODES * C_DIM * 4;  // 12.29 MB
  float* skipb = (float*)ws;  ws += (size_t)N_NODES * C_DIM * 4;  // 12.29 MB
  float* ssrc  = (float*)ws;  ws += (size_t)NH * N_NODES * 4;     // 192 KB
  float* stgt  = (float*)ws;  ws += (size_t)NH * JPAD * 4;        // 193 KB
  __bf16* projT = (__bf16*)ws;                                    // 6.16 MB

  const dim3 gemmBlk(32, 4);
  const dim3 gemmGrd(8, 375);
  gat_proj_gemm<<<gemmGrd, gemmBlk, 0, stream>>>(x, Wm, proj);
  gat_proj_gemm<<<gemmGrd, gemmBlk, 0, stream>>>(x, skW, skipb);

  gat_scores<<<(NH * JPAD + 255) / 256, 256, 0, stream>>>(proj, a_src, a_tgt,
                                                          ssrc, stgt);
  gat_transpose_bf16<<<dim3(JPAD / 32, C_DIM / 32), dim3(32, 8), 0, stream>>>(
      proj, projT);
  gat_flash_aggregate<<<375, dim3(32, NH), 0, stream>>>(adj, ssrc, stgt, projT,
                                                        skipb, bias, out);
}
